// TemporalBlock_87840671138376
// MI455X (gfx1250) — compile-verified, matrix operand order checked by reading, not run on hardware
//
#include <hip/hip_runtime.h>
#include <math.h>

typedef __bf16 bf16;
typedef __attribute__((ext_vector_type(16))) __bf16 v16bf;
typedef __attribute__((ext_vector_type(8)))  float   v8f;

#define BB   4
#define CIN  256
#define COUT 512
#define SS   1024
#define KSZ  3
#define PADL 2
#define NH   8
#define HD   64
#define NTOK (BB * SS)     /* 4096 */
#define SPAD (SS + PADL)   /* 1026 */

__device__ __forceinline__ int lane_id() { return threadIdx.x & 31; }

__device__ __forceinline__ int frag_kbase(int i, int hi) {
  // bf16 A/B fragment K layout per CDNA5 ISA 7.12.2 (16-bit A 16x32):
  // VGPR0..3: K = hi*8 + 2i ; VGPR4..7: K = 16 + hi*8 + 2(i-4)
  return (i < 4) ? (hi * 8 + 2 * i) : (16 + hi * 8 + 2 * (i - 4));
}

// Fragment from a row-major M x K tile (also serves B given as N x K = B^T).
__device__ __forceinline__ v16bf load_frag_nk(const bf16* __restrict__ p, int ld) {
  const int l = lane_id(), m = l & 15, hi = l >> 4;
  v16bf f;
#pragma unroll
  for (int i = 0; i < 8; ++i) {
    const int kb = frag_kbase(i, hi);
    f[2 * i]     = p[m * ld + kb];
    f[2 * i + 1] = p[m * ld + kb + 1];
  }
  return f;
}

// B fragment from a row-major K x N tile (K = reduction dim).
__device__ __forceinline__ v16bf load_frag_kn(const bf16* __restrict__ p, int ld) {
  const int l = lane_id(), n = l & 15, hi = l >> 4;
  v16bf f;
#pragma unroll
  for (int i = 0; i < 8; ++i) {
    const int kb = frag_kbase(i, hi);
    f[2 * i]     = p[kb * ld + n];
    f[2 * i + 1] = p[(kb + 1) * ld + n];
  }
  return f;
}

__device__ __forceinline__ v8f wmma_bf16(v16bf a, v16bf b, v8f c) {
  return __builtin_amdgcn_wmma_f32_16x16x32_bf16(false, a, false, b, (short)0, c,
                                                 false, false);
}

__device__ __forceinline__ v8f zero8() {
  v8f z;
#pragma unroll
  for (int r = 0; r < 8; ++r) z[r] = 0.0f;
  return z;
}

__device__ __forceinline__ float wred_max(float v) {
#pragma unroll
  for (int o = 16; o > 0; o >>= 1) v = fmaxf(v, __shfl_xor(v, o, 32));
  return v;
}
__device__ __forceinline__ float wred_sum(float v) {
#pragma unroll
  for (int o = 16; o > 0; o >>= 1) v += __shfl_xor(v, o, 32);
  return v;
}

// CDNA5 async global -> LDS copy, 16 bytes per lane (ASYNCcnt-tracked).
__device__ __forceinline__ void async_copy16(const bf16* gsrc, bf16* lds_dst) {
  const unsigned long long ga = (unsigned long long)gsrc;
  const unsigned loff = (unsigned)(unsigned long long)lds_dst;  // LDS byte offset
  asm volatile("global_load_async_to_lds_b128 %0, %1, off"
               :
               : "v"(loff), "v"(ga)
               : "memory");
}
__device__ __forceinline__ void wait_async0() {
  asm volatile("s_wait_asynccnt 0x0" ::: "memory");
}

// ---------------------------------------------------------------------------
// Generic bf16 WMMA GEMM: C(M=512 x N=4096) = A(M x K) * Bt(N x K)^T
// Double-buffered async-to-LDS staging; epilogue y = acc*scale[m]+shift[m].
// ---------------------------------------------------------------------------
#define TM 128
#define TN 64
#define TK 32

// OUTMODE: 0 f32 [n*COUT+m] | 1 f32+bf16 [n*COUT+m] | 2 bf16 conv-pad | 3 bf16 head
template <int OUTMODE, bool RELU>
__global__ __launch_bounds__(256) void gemm_kernel(
    const bf16* __restrict__ A, const bf16* __restrict__ Bt, int K,
    const float* __restrict__ scale, const float* __restrict__ shift,
    float* __restrict__ Cf, bf16* __restrict__ Cb) {
  __shared__ __align__(16) bf16 As[2][TM * TK];
  __shared__ __align__(16) bf16 Bs[2][TN * TK];
  const int tid = threadIdx.x;
  const int wave = tid >> 5;
  const int m0 = blockIdx.x * TM;
  const int n0 = blockIdx.y * TN;
  const int wm = wave >> 1, wn = wave & 1;  // 4x2 waves, 32x32 each

  v8f acc[2][2];
#pragma unroll
  for (int i = 0; i < 2; ++i)
#pragma unroll
    for (int j = 0; j < 2; ++j) acc[i][j] = zero8();

  auto issue_tile = [&](int buf, int k0) {
    for (int idx = tid; idx < (TM * TK) / 8; idx += 256) {
      const int row = idx >> 2, c = (idx & 3) * 8;
      async_copy16(&A[(size_t)(m0 + row) * K + k0 + c], &As[buf][row * TK + c]);
    }
    for (int idx = tid; idx < (TN * TK) / 8; idx += 256) {
      const int row = idx >> 2, c = (idx & 3) * 8;
      async_copy16(&Bt[(size_t)(n0 + row) * K + k0 + c], &Bs[buf][row * TK + c]);
    }
  };

  issue_tile(0, 0);
  int cur = 0;
  for (int k0 = 0; k0 < K; k0 += TK) {
    wait_async0();       // this wave's async copies into As/Bs[cur] are done
    __syncthreads();     // whole tile visible; prior compute on buf^1 finished
    if (k0 + TK < K) issue_tile(cur ^ 1, k0 + TK);
    v16bf bfr[2];
#pragma unroll
    for (int j = 0; j < 2; ++j)
      bfr[j] = load_frag_nk(&Bs[cur][(wn * 32 + j * 16) * TK], TK);
#pragma unroll
    for (int i = 0; i < 2; ++i) {
      v16bf af = load_frag_nk(&As[cur][(wm * 32 + i * 16) * TK], TK);
#pragma unroll
      for (int j = 0; j < 2; ++j) acc[i][j] = wmma_bf16(af, bfr[j], acc[i][j]);
    }
    cur ^= 1;
  }

  const int l = lane_id(), nl = l & 15, hi = l >> 4;
#pragma unroll
  for (int i = 0; i < 2; ++i)
#pragma unroll
    for (int j = 0; j < 2; ++j) {
      const int mt = m0 + wm * 32 + i * 16;
      const int nt = n0 + wn * 32 + j * 16 + nl;
#pragma unroll
      for (int r = 0; r < 8; ++r) {
        const int m = mt + r + hi * 8;
        float y = acc[i][j][r] * scale[m] + shift[m];
        if (RELU) y = fmaxf(y, 0.0f);
        if (OUTMODE == 0) {
          Cf[(size_t)nt * COUT + m] = y;
        } else if (OUTMODE == 1) {
          Cf[(size_t)nt * COUT + m] = y;
          Cb[(size_t)nt * COUT + m] = (bf16)y;
        } else if (OUTMODE == 2) {
          const int b = nt >> 10, s = nt & (SS - 1);
          Cb[((size_t)b * COUT + m) * SPAD + PADL + s] = (bf16)y;
        } else {
          const int b = nt >> 10, s = nt & (SS - 1);
          const int h = m >> 6, d = m & 63;
          Cb[(((size_t)b * NH + h) * SS + s) * HD + d] = (bf16)y;
        }
      }
    }
}

// ---------------------------------------------------------------------------
// Attention: one workgroup (16 waves) per (b, h, 16-query tile).
// scores kept entirely in LDS; double softmax + decay; WMMA for QK^T and PV.
// ---------------------------------------------------------------------------
#define ATT_SMEM (16 * 1024 * 4 + 16 * 1024 * 2 + 16 * 64 * 2)

__global__ __launch_bounds__(512) void attention_kernel(
    const bf16* __restrict__ Q, const bf16* __restrict__ Kc,
    const bf16* __restrict__ V, const float* __restrict__ gdev,
    bf16* __restrict__ AO) {
  extern __shared__ char smem[];
  float* sc = (float*)smem;                                     // 16 x 1024 f32
  bf16* aw  = (bf16*)(smem + 16 * 1024 * 4);                    // 16 x 1024 bf16
  bf16* qt  = (bf16*)(smem + 16 * 1024 * 4 + 16 * 1024 * 2);    // 16 x 64 bf16

  const int tid = threadIdx.x;
  const int wave = tid >> 5;
  const int l = lane_id();
  const int bid = blockIdx.x;
  const int qtile = bid & 63;
  const int h = (bid >> 6) & 7;
  const int b = bid >> 9;
  const int q0 = qtile * 16;
  const size_t hb = ((size_t)b * NH + h) * SS;
  const bf16* Qp = Q + hb * HD;
  const bf16* Kp = Kc + hb * HD;
  const bf16* Vp = V + hb * HD;

  for (int idx = tid; idx < 16 * HD; idx += 512)
    qt[idx] = Qp[(size_t)(q0 + (idx >> 6)) * HD + (idx & 63)];
  __syncthreads();

  // Scores: Q(16x64) x K^T(64x1024); wave w handles key columns [w*64, w*64+64)
  {
    v16bf a0 = load_frag_nk(&qt[0], HD);
    v16bf a1 = load_frag_nk(&qt[32], HD);
    const int nw = wave * 64;
#pragma unroll
    for (int j = 0; j < 4; ++j) {
      const int n0 = nw + j * 16;
      v8f acc = zero8();
      v16bf b0 = load_frag_nk(Kp + (size_t)n0 * HD, HD);
      acc = wmma_bf16(a0, b0, acc);
      v16bf b1 = load_frag_nk(Kp + (size_t)n0 * HD + 32, HD);
      acc = wmma_bf16(a1, b1, acc);
      const int nl = l & 15, hi = l >> 4;
#pragma unroll
      for (int r = 0; r < 8; ++r)
        sc[(r + hi * 8) * 1024 + n0 + nl] = acc[r] * 0.125f;  // 1/sqrt(64)
    }
  }
  __syncthreads();

  // Row processing: wave w owns query row w.
  {
    float* srow = sc + wave * 1024;
    bf16* arow = aw + wave * 1024;
    const float g = gdev[h];
    const int q = q0 + wave;

    float mx = -3.0e38f;
    for (int c = l; c < 1024; c += 32) mx = fmaxf(mx, srow[c]);
    mx = wred_max(mx);
    float sum = 0.0f;
    for (int c = l; c < 1024; c += 32) sum += __expf(srow[c] - mx);
    sum = wred_sum(sum);
    const float invs = 1.0f / sum;

    // Ordered pass: cumsum of softmax probs via wave inclusive scan,
    // decay-modulated scores written back in place. disttotal == 1 exactly.
    float running = 0.0f, mx2 = -3.0e38f;
    for (int base = 0; base < 1024; base += 32) {
      const float v = srow[base + l];
      const float p = __expf(v - mx) * invs;
      float incl = p;
#pragma unroll
      for (int o = 1; o < 32; o <<= 1) {
        const float t = __shfl_up(incl, o, 32);
        if (l >= o) incl += t;
      }
      const float cum = running + incl;
      running += __shfl(incl, 31, 32);
      const float pos = fabsf((float)(q - (base + l)));
      const float ds = sqrtf(fmaxf((1.0f - cum) * pos, 0.0f));
      const float eff = fminf(fmaxf(__expf(ds * g), 1e-5f), 1e5f);
      const float ns = v * eff;
      srow[base + l] = ns;
      mx2 = fmaxf(mx2, ns);
    }
    mx2 = wred_max(mx2);
    float sum2 = 0.0f;
    for (int c = l; c < 1024; c += 32) sum2 += __expf(srow[c] - mx2);
    sum2 = wred_sum(sum2);
    const float inv2 = 1.0f / sum2;
    for (int c = l; c < 1024; c += 32) {
      const float pv = (q == 0) ? 0.0f : __expf(srow[c] - mx2) * inv2;
      arow[c] = (bf16)pv;
    }
  }
  __syncthreads();

  // AO tile: aw2(16x1024) x V(1024x64); waves 0..3 take d-columns.
  if (wave < 4) {
    const int n0 = wave * 16;
    v8f acc = zero8();
    for (int kt = 0; kt < 1024; kt += 32) {
      v16bf a = load_frag_nk(&aw[kt], 1024);
      v16bf bv = load_frag_kn(Vp + (size_t)kt * HD + n0, HD);
      acc = wmma_bf16(a, bv, acc);
    }
    const int nl = l & 15, hi = l >> 4;
#pragma unroll
    for (int r = 0; r < 8; ++r) {
      const int m = r + hi * 8;
      AO[((size_t)b * SS + q0 + m) * COUT + h * HD + n0 + nl] = (bf16)acc[r];
    }
  }
}

// ---------------------------------------------------------------------------
// Elementwise / prep kernels
// ---------------------------------------------------------------------------
__global__ void prep_kernel(const float* c1b, const float* g1, const float* b1,
                            const float* m1, const float* v1, const float* c2b,
                            const float* g2, const float* b2, const float* m2,
                            const float* v2, const float* gamma, float* scale1,
                            float* shift1, float* scale2, float* shift2,
                            float* ones, float* gdev) {
  const int i = threadIdx.x;
  if (i < COUT) {
    const float s1 = g1[i] * rsqrtf(v1[i] + 1e-5f);
    scale1[i] = s1;
    shift1[i] = (c1b[i] - m1[i]) * s1 + b1[i];
    const float s2 = g2[i] * rsqrtf(v2[i] + 1e-5f);
    scale2[i] = s2;
    shift2[i] = (c2b[i] - m2[i]) * s2 + b2[i];
    ones[i] = 1.0f;
  }
  if (i < NH) gdev[i] = -log1pf(__expf(gamma[i]));  // -softplus(gamma)
}

__global__ void f2b_kernel(const float* __restrict__ s, bf16* __restrict__ d,
                           int n) {
  const int i = blockIdx.x * blockDim.x + threadIdx.x;
  if (i < n) d[i] = (bf16)s[i];
}

// d[m*512 + k] = s[k*512 + m]   (transpose 512x512, f32 -> bf16)
__global__ void f2bT512_kernel(const float* __restrict__ s,
                               bf16* __restrict__ d) {
  const int i = blockIdx.x * blockDim.x + threadIdx.x;
  if (i < COUT * COUT) d[i] = (bf16)s[(i & 511) * COUT + (i >> 9)];
}

__global__ void build_xpad_kernel(const float* __restrict__ x,
                                  bf16* __restrict__ xp) {
  const int i = blockIdx.x * blockDim.x + threadIdx.x;
  if (i >= BB * CIN * SPAD) return;
  const int t = i % SPAD;
  const int bc = i / SPAD;
  xp[i] = (t < PADL) ? (bf16)(-1.0f) : (bf16)x[(size_t)bc * SS + (t - PADL)];
}

__global__ void fill_hpad_kernel(bf16* __restrict__ hp) {
  const int i = blockIdx.x * blockDim.x + threadIdx.x;
  if (i >= BB * COUT * PADL) return;
  const int bc = i / PADL, t = i % PADL;
  hp[(size_t)bc * SPAD + t] = (bf16)(-1.0f);
}

// xT[(b*S+s)*CIN + c] = x[(b*CIN+c)*S + s]
__global__ void build_xT_kernel(const float* __restrict__ x,
                                bf16* __restrict__ xT) {
  const int i = blockIdx.x * blockDim.x + threadIdx.x;
  if (i >= NTOK * CIN) return;
  const int c = i & (CIN - 1);
  const int n = i >> 8;
  const int b = n >> 10, s = n & 1023;
  xT[i] = (bf16)x[((size_t)b * CIN + c) * SS + s];
}

// col[(b*S+s)*(C*3) + ci*3+kk] = pad[(b*C+ci)*SPAD + s + kk]
__global__ void im2col_kernel(const bf16* __restrict__ pad,
                              bf16* __restrict__ col, int C) {
  const int Kc = C * KSZ;
  const size_t total = (size_t)NTOK * Kc;
  for (size_t i = (size_t)blockIdx.x * blockDim.x + threadIdx.x; i < total;
       i += (size_t)gridDim.x * blockDim.x) {
    const int k = (int)(i % Kc);
    const int n = (int)(i / Kc);
    const int b = n >> 10, s = n & 1023;
    const int ci = k / KSZ, kk = k - ci * KSZ;
    col[i] = pad[((size_t)b * C + ci) * SPAD + s + kk];
  }
}

// ---------------------------------------------------------------------------
// Final: two residual LayerNorms + ReLU + transpose to (B, C, S)
// ---------------------------------------------------------------------------
__device__ __forceinline__ float block_sum256(float v, float* sred) {
  v = wred_sum(v);
  if ((threadIdx.x & 31) == 0) sred[threadIdx.x >> 5] = v;
  __syncthreads();
  float r = (threadIdx.x < 8) ? sred[threadIdx.x] : 0.0f;
  r = wred_sum(r);
  if (threadIdx.x == 0) sred[0] = r;
  __syncthreads();
  const float out = sred[0];
  __syncthreads();
  return out;
}

__global__ __launch_bounds__(256) void final_kernel(
    const float* __restrict__ xt, const float* __restrict__ aop,
    const float* __restrict__ res, const float* __restrict__ g1,
    const float* __restrict__ b1, const float* __restrict__ g2,
    const float* __restrict__ b2, float* __restrict__ out) {
  __shared__ float sred[8];
  const int n = blockIdx.x;
  const int b = n >> 10, s = n & 1023;
  const int t = threadIdx.x;
  const float* xr = xt + (size_t)n * COUT;
  const float* ar = aop + (size_t)n * COUT;
  const float* rr = res + (size_t)n * COUT;

  const float v0 = xr[t] + ar[t];
  const float v1 = xr[t + 256] + ar[t + 256];
  const float mu = block_sum256(v0 + v1, sred) * (1.0f / COUT);
  const float d0 = v0 - mu, d1 = v1 - mu;
  const float var = block_sum256(d0 * d0 + d1 * d1, sred) * (1.0f / COUT);
  const float rs = rsqrtf(var + 1e-5f);
  const float a0 = d0 * rs * g1[t] + b1[t];
  const float a1 = d1 * rs * g1[t + 256] + b1[t + 256];

  const float w0 = xr[t] + a0, w1 = xr[t + 256] + a1;
  const float mu2 = block_sum256(w0 + w1, sred) * (1.0f / COUT);
  const float e0 = w0 - mu2, e1 = w1 - mu2;
  const float var2 = block_sum256(e0 * e0 + e1 * e1, sred) * (1.0f / COUT);
  const float rs2 = rsqrtf(var2 + 1e-5f);

  const float y0 = fmaxf(e0 * rs2 * g2[t] + b2[t] + rr[t], 0.0f);
  const float y1 = fmaxf(e1 * rs2 * g2[t + 256] + b2[t + 256] + rr[t + 256], 0.0f);
  float* ob = out + (size_t)b * COUT * SS;
  ob[(size_t)t * SS + s] = y0;
  ob[(size_t)(t + 256) * SS + s] = y1;
}

// ---------------------------------------------------------------------------
extern "C" void kernel_launch(void* const* d_in, const int* in_sizes, int n_in,
                              void* d_out, int out_size, void* d_ws,
                              size_t ws_size, hipStream_t stream) {
  (void)in_sizes; (void)n_in; (void)out_size; (void)ws_size;
  const float* x    = (const float*)d_in[0];
  const float* c1w  = (const float*)d_in[1];
  const float* c1b  = (const float*)d_in[2];
  const float* bn1g = (const float*)d_in[3];
  const float* bn1b = (const float*)d_in[4];
  const float* bn1m = (const float*)d_in[5];
  const float* bn1v = (const float*)d_in[6];
  const float* c2w  = (const float*)d_in[7];
  const float* c2b  = (const float*)d_in[8];
  const float* bn2g = (const float*)d_in[9];
  const float* bn2b = (const float*)d_in[10];
  const float* bn2m = (const float*)d_in[11];
  const float* bn2v = (const float*)d_in[12];
  const float* dww  = (const float*)d_in[13];
  const float* dwb  = (const float*)d_in[14];
  const float* qw   = (const float*)d_in[15];
  const float* qb   = (const float*)d_in[16];
  const float* kw   = (const float*)d_in[17];
  const float* kb   = (const float*)d_in[18];
  const float* vw   = (const float*)d_in[19];
  const float* vb   = (const float*)d_in[20];
  const float* ow   = (const float*)d_in[21];
  const float* ob   = (const float*)d_in[22];
  const float* gamma = (const float*)d_in[23];
  const float* ln1g = (const float*)d_in[24];
  const float* ln1b = (const float*)d_in[25];
  const float* ln2g = (const float*)d_in[26];
  const float* ln2b = (const float*)d_in[27];

  char* ws = (char*)d_ws;
  size_t off = 0;
  auto alloc = [&](size_t bytes) -> void* {
    void* p = ws + off;
    off = (off + bytes + 255) & ~(size_t)255;
    return p;
  };

  bf16* xpad = (bf16*)alloc((size_t)BB * CIN * SPAD * 2);
  bf16* hpad = (bf16*)alloc((size_t)BB * COUT * SPAD * 2);
  bf16* A1   = (bf16*)alloc((size_t)COUT * CIN * KSZ * 2);
  bf16* A2   = (bf16*)alloc((size_t)COUT * COUT * KSZ * 2);
  bf16* Ad   = (bf16*)alloc((size_t)COUT * CIN * 2);
  bf16* Aq   = (bf16*)alloc((size_t)COUT * COUT * 2);
  bf16* Ak   = (bf16*)alloc((size_t)COUT * COUT * 2);
  bf16* Av   = (bf16*)alloc((size_t)COUT * COUT * 2);
  bf16* Aow  = (bf16*)alloc((size_t)COUT * COUT * 2);
  bf16* col  = (bf16*)alloc((size_t)NTOK * COUT * KSZ * 2);  // reused conv1/conv2
  bf16* xT   = (bf16*)alloc((size_t)NTOK * CIN * 2);
  float* xt  = (float*)alloc((size_t)NTOK * COUT * 4);
  bf16* xtb  = (bf16*)alloc((size_t)NTOK * COUT * 2);
  float* resb = (float*)alloc((size_t)NTOK * COUT * 4);
  bf16* Qh   = (bf16*)alloc((size_t)NTOK * COUT * 2);
  bf16* Kh   = (bf16*)alloc((size_t)NTOK * COUT * 2);
  bf16* Vh   = (bf16*)alloc((size_t)NTOK * COUT * 2);
  bf16* AO   = (bf16*)alloc((size_t)NTOK * COUT * 2);
  float* aop = (float*)alloc((size_t)NTOK * COUT * 4);
  float* scale1 = (float*)alloc(COUT * 4);
  float* shift1 = (float*)alloc(COUT * 4);
  float* scale2 = (float*)alloc(COUT * 4);
  float* shift2 = (float*)alloc(COUT * 4);
  float* ones   = (float*)alloc(COUT * 4);
  float* gdev   = (float*)alloc(64 * 4);

  const dim3 ggrid(COUT / TM, NTOK / TN);

  prep_kernel<<<1, 512, 0, stream>>>(c1b, bn1g, bn1b, bn1m, bn1v, c2b, bn2g,
                                     bn2b, bn2m, bn2v, gamma, scale1, shift1,
                                     scale2, shift2, ones, gdev);
  f2b_kernel<<<(COUT * CIN * KSZ + 255) / 256, 256, 0, stream>>>(c1w, A1,
                                                                 COUT * CIN * KSZ);
  f2b_kernel<<<(COUT * COUT * KSZ + 255) / 256, 256, 0, stream>>>(c2w, A2,
                                                                  COUT * COUT * KSZ);
  f2b_kernel<<<(COUT * CIN + 255) / 256, 256, 0, stream>>>(dww, Ad, COUT * CIN);
  f2bT512_kernel<<<(COUT * COUT) / 256, 256, 0, stream>>>(qw, Aq);
  f2bT512_kernel<<<(COUT * COUT) / 256, 256, 0, stream>>>(kw, Ak);
  f2bT512_kernel<<<(COUT * COUT) / 256, 256, 0, stream>>>(vw, Av);
  f2bT512_kernel<<<(COUT * COUT) / 256, 256, 0, stream>>>(ow, Aow);
  build_xpad_kernel<<<(BB * CIN * SPAD + 255) / 256, 256, 0, stream>>>(x, xpad);
  build_xT_kernel<<<(NTOK * CIN) / 256, 256, 0, stream>>>(x, xT);
  fill_hpad_kernel<<<(BB * COUT * PADL + 255) / 256, 256, 0, stream>>>(hpad);

  // conv1 (+BN+ReLU) as GEMM: M=512, N=4096, K=768 -> padded bf16 h
  im2col_kernel<<<2048, 256, 0, stream>>>(xpad, col, CIN);
  gemm_kernel<2, true><<<ggrid, 256, 0, stream>>>(A1, col, CIN * KSZ, scale1,
                                                  shift1, nullptr, hpad);
  // conv2 (+BN+ReLU): K=1536 -> xt (f32, (B,S,E)) + xtb (bf16)
  im2col_kernel<<<4096, 256, 0, stream>>>(hpad, col, COUT);
  gemm_kernel<1, true><<<ggrid, 256, 0, stream>>>(A2, col, COUT * KSZ, scale2,
                                                  shift2, xt, xtb);
  // 1x1 downsample: K=256 -> res (f32, (B,S,E))
  gemm_kernel<0, false><<<ggrid, 256, 0, stream>>>(Ad, xT, CIN, ones, dwb, resb,
                                                   nullptr);
  // Q/K/V projections -> head layout bf16 (B,H,S,HD)
  gemm_kernel<3, false><<<ggrid, 256, 0, stream>>>(Aq, xtb, COUT, ones, qb,
                                                   nullptr, Qh);
  gemm_kernel<3, false><<<ggrid, 256, 0, stream>>>(Ak, xtb, COUT, ones, kb,
                                                   nullptr, Kh);
  gemm_kernel<3, false><<<ggrid, 256, 0, stream>>>(Av, xtb, COUT, ones, vb,
                                                   nullptr, Vh);
  // attention: 4*8*64 = 2048 workgroups, scores resident in LDS
  attention_kernel<<<BB * NH * (SS / 16), 512, ATT_SMEM, stream>>>(Qh, Kh, Vh,
                                                                   gdev, AO);
  // output projection -> aop (f32)
  gemm_kernel<0, false><<<ggrid, 256, 0, stream>>>(Aow, AO, COUT, ones, ob, aop,
                                                   nullptr);
  // two residual LayerNorms + ReLU, transposed store to (B,C,S)
  final_kernel<<<NTOK, 256, 0, stream>>>(xt, aop, resb, ln1g, ln1b, ln2g, ln2b,
                                         (float*)d_out);
}
